// EGNNLayer_73804718015038
// MI455X (gfx1250) — matrix operand order repaired and checked by reading, not compiled, not run on hardware
//
#include <hip/hip_runtime.h>

// ---------------------------------------------------------------------------
// EGNN layer for MI455X (gfx1250): bf16 WMMA edge/node MLPs + atomic scatter.
// Single manually-partitioned LDS block: weights at offset 0 so ds_load
// offsets fold as immediates; per-wave tiles last. B frags software-prefetched.
// ---------------------------------------------------------------------------

typedef __attribute__((ext_vector_type(16))) __bf16       v16bf;
typedef __attribute__((ext_vector_type(8)))  float        v8f;
typedef __attribute__((ext_vector_type(4)))  unsigned int u32x4;
typedef __attribute__((ext_vector_type(2)))  float        f32x2;

union FragBF { v16bf v; u32x4 q[2]; };

__device__ __forceinline__ unsigned short f2bf(float f) {
  union { float f; unsigned u; } c; c.f = f;
  unsigned u = c.u;
  u += 0x7FFFu + ((u >> 16) & 1u);       // round-to-nearest-even
  return (unsigned short)(u >> 16);
}

__device__ __forceinline__ unsigned pack2bf(float lo, float hi) {
  return (unsigned)f2bf(lo) | ((unsigned)f2bf(hi) << 16);
}

// A-matrix 16x32 bf16 fragment from LDS row pointer p = &row[ktile*32 + 8*h].
// Elements e=0..7 hold k = base..base+7, e=8..15 hold k = base+16..base+23.
__device__ __forceinline__ v16bf frag_a_lds(const unsigned short* p) {
  FragBF f;
  f.q[0] = *(const u32x4*)(p);
  f.q[1] = *(const u32x4*)(p + 16);
  return f.v;
}

// B-matrix fragment: 32 contiguous bytes per lane (pre-swizzled layout).
__device__ __forceinline__ v16bf frag_b(const unsigned short* p) {
  FragBF f;
  f.q[0] = *(const u32x4*)(p);
  f.q[1] = *(const u32x4*)(p + 8);
  return f.v;
}

__device__ __forceinline__ float silu_f(float v) {
  return v / (1.0f + __expf(-v));
}

__device__ __forceinline__ v8f vzero8() {
  v8f z = {0.f, 0.f, 0.f, 0.f, 0.f, 0.f, 0.f, 0.f};
  return z;
}

#define LDA 168           // LDS row stride (ushort), edge-input tile (160 used)
#define LDH 136           // LDS row stride, hidden tile (128 used)
#define LDN 200           // LDS row stride, node-input tile (192 used)
#define TILE_A (16 * LDA) // 2688 ushorts
#define TILE_H (16 * LDH) // 2176 ushorts
#define TILE_N (16 * LDN) // 3200 ushorts

// Edge-kernel LDS layout (ushort offsets). W2 first: all layer-2 frags fold.
#define EW2_OFF 0                      // 4*8*512 = 16384 ushorts (32 KB)
#define EW1_OFF 16384                  // 5*8*512 = 20480 ushorts (40 KB)
#define ES_OFF  (16384 + 20480)        // per-wave tiles
#define E_LDS   (ES_OFF + 8 * 2 * TILE_A)

// Node-kernel LDS layout. Total weight region = exactly 64 KB: all fold.
#define NW2_OFF 0                      // 4*4*512 =  8192 ushorts (16 KB)
#define NW1_OFF 8192                   // 6*8*512 = 24576 ushorts (48 KB)
#define NS_OFF  (8192 + 24576)
#define N_LDS   (NS_OFF + 8 * 2 * TILE_N)

// ---------------------------------------------------------------------------
// Weight pre-swizzle: fp32 [K x Nsrc] row-major  ->  bf16 WMMA B fragments.
// Fragment (kt, nt): 32 lanes x 16 bf16 contiguous.
//   n = nt*16 + (lane & 15);  k = kt*32 + (e>>3)*16 + (lane>>4)*8 + (e&7)
// ---------------------------------------------------------------------------
__global__ void swz_weights(const float* __restrict__ W, unsigned short* __restrict__ dst,
                            int Kact, int KT, int NT, int Nsrc) {
  int total = KT * NT * 32 * 16;
  for (int t = blockIdx.x * blockDim.x + threadIdx.x; t < total;
       t += gridDim.x * blockDim.x) {
    int e    = t & 15;
    int l    = (t >> 4) & 31;
    int fidx = t >> 9;
    int nt   = fidx % NT;
    int kt   = fidx / NT;
    int n    = nt * 16 + (l & 15);
    int k    = kt * 32 + ((e & 8) << 1) + ((l >> 4) << 3) + (e & 7);
    float v  = (k < Kact && n < Nsrc) ? W[k * Nsrc + n] : 0.0f;
    dst[t]   = f2bf(v);
  }
}

__global__ void zero_f32v4(float4* __restrict__ p, int n4) {
  int i = blockIdx.x * blockDim.x + threadIdx.x;
  if (i < n4) p[i] = make_float4(0.f, 0.f, 0.f, 0.f);
}

// ---------------------------------------------------------------------------
// Edge kernel: per wave, 32 edges (2 M-tiles sharing every B fragment).
//   edge_input[32x145] -> silu(.@ew1+eb1) -> silu(.@ew2+eb2) -> atomic scatter
// ---------------------------------------------------------------------------
__global__ void __launch_bounds__(256)
edge_mlp_scatter(const float* __restrict__ x,
                 const int*   __restrict__ ei,     // [2][E]
                 const float* __restrict__ ewt,    // [E]
                 const float* __restrict__ eattr,  // [E][16]
                 const unsigned short* __restrict__ w1f,  // 5x8 frags (global)
                 const float* __restrict__ b1,
                 const unsigned short* __restrict__ w2f,  // 4x8 frags (global)
                 const float* __restrict__ b2,
                 float* __restrict__ agg,          // [N][128]
                 int E) {
  __shared__ __align__(16) unsigned short L[E_LDS];

  const int lane = threadIdx.x & 31;
  const int wv   = threadIdx.x >> 5;
  const int e0   = blockIdx.x * 256 + wv * 32;
  unsigned short* buf = &L[ES_OFF + wv * (2 * TILE_A)];

  const int m = lane & 15;   // A row / D column index
  const int h = lane >> 4;   // lane half

  // ---- stage 0: cooperatively stage weight fragments into LDS ----
  {
    const u32x4* s1 = (const u32x4*)w1f;
    const u32x4* s2 = (const u32x4*)w2f;
    u32x4* d1 = (u32x4*)&L[EW1_OFF];
    u32x4* d2 = (u32x4*)&L[EW2_OFF];
    for (int i = threadIdx.x; i < 5 * 8 * 512 / 8; i += 256) d1[i] = s1[i];
    for (int i = threadIdx.x; i < 4 * 8 * 512 / 8; i += 256) d2[i] = s2[i];
  }

  // ---- stage 1: gather edge_input tiles [32 x 160] (bf16 pairs) ----
  for (int e = 0; e < 32; ++e) {
    int ge = e0 + e; if (ge >= E) ge = E - 1;
    int src0 = ei[ge];
    int src1 = ei[E + ge];
    unsigned* dst = (unsigned*)&buf[(e >> 4) * TILE_A + (e & 15) * LDA];
    #pragma unroll
    for (int j = 0; j < 3; ++j) {
      int p = (j << 5) + lane;          // column-pair index, valid < 80
      if (p < 80) {
        int c = p << 1;
        float v0, v1;
        if (c < 64)        { f32x2 t = *(const f32x2*)&x[src0 * 64 + c];        v0 = t.x; v1 = t.y; }
        else if (c < 128)  { f32x2 t = *(const f32x2*)&x[src1 * 64 + (c - 64)]; v0 = t.x; v1 = t.y; }
        else if (c < 144)  { f32x2 t = __builtin_nontemporal_load(
                                 (const f32x2*)&eattr[ge * 16 + (c - 128)]);    v0 = t.x; v1 = t.y; }
        else if (c == 144) { v0 = __builtin_nontemporal_load(&ewt[ge]) * 0.2f;  v1 = 0.f; }
        else               { v0 = 0.f; v1 = 0.f; }        // K padding 146..159
        dst[p] = pack2bf(v0, v1);
      }
    }
  }

  __syncthreads();   // weights staged; per-wave tiles need no barrier

  const unsigned short* wlane = &L[lane << 4];   // per-lane weight base

  // ---- stage 2: layer 1 (K=160 -> 5 ktiles, N=128 -> 8 ntiles) ----
  v8f acc0[8], acc1[8];
  #pragma unroll
  for (int nt = 0; nt < 8; ++nt) { acc0[nt] = vzero8(); acc1[nt] = vzero8(); }
  #pragma unroll
  for (int kt = 0; kt < 5; ++kt) {
    v16bf a0 = frag_a_lds(&buf[m * LDA + kt * 32 + (h << 3)]);
    v16bf a1 = frag_a_lds(&buf[TILE_A + m * LDA + kt * 32 + (h << 3)]);
    v16bf bc = frag_b(wlane + EW1_OFF + (kt << 12));
    #pragma unroll
    for (int nt = 0; nt < 8; ++nt) {
      v16bf bn;
      if (nt < 7) bn = frag_b(wlane + EW1_OFF + (kt << 12) + ((nt + 1) << 9));
      acc0[nt] = __builtin_amdgcn_wmma_f32_16x16x32_bf16(
          false, a0, false, bc, (short)0, acc0[nt], false, false);
      acc1[nt] = __builtin_amdgcn_wmma_f32_16x16x32_bf16(
          false, a1, false, bc, (short)0, acc1[nt], false, false);
      if (nt < 7) bc = bn;
    }
  }
  // bias + silu -> hidden tiles (bf16), overlaying the input tiles
  #pragma unroll
  for (int nt = 0; nt < 8; ++nt) {
    float bias = b1[(nt << 4) + m];
    #pragma unroll
    for (int r = 0; r < 8; ++r) {
      float v0 = silu_f(acc0[nt][r] + bias);
      float v1 = silu_f(acc1[nt][r] + bias);
      buf[(r + (h << 3)) * LDH + (nt << 4) + m]          = f2bf(v0);
      buf[TILE_H + (r + (h << 3)) * LDH + (nt << 4) + m] = f2bf(v1);
    }
  }

  // ---- stage 3: layer 2 (K=128 -> 4 ktiles, N=128 -> 8 ntiles) ----
  #pragma unroll
  for (int nt = 0; nt < 8; ++nt) { acc0[nt] = vzero8(); acc1[nt] = vzero8(); }
  #pragma unroll
  for (int kt = 0; kt < 4; ++kt) {
    v16bf a0 = frag_a_lds(&buf[m * LDH + kt * 32 + (h << 3)]);
    v16bf a1 = frag_a_lds(&buf[TILE_H + m * LDH + kt * 32 + (h << 3)]);
    v16bf bc = frag_b(wlane + EW2_OFF + (kt << 12));
    #pragma unroll
    for (int nt = 0; nt < 8; ++nt) {
      v16bf bn;
      if (nt < 7) bn = frag_b(wlane + EW2_OFF + (kt << 12) + ((nt + 1) << 9));
      acc0[nt] = __builtin_amdgcn_wmma_f32_16x16x32_bf16(
          false, a0, false, bc, (short)0, acc0[nt], false, false);
      acc1[nt] = __builtin_amdgcn_wmma_f32_16x16x32_bf16(
          false, a1, false, bc, (short)0, acc1[nt], false, false);
      if (nt < 7) bc = bn;
    }
  }

  // ---- stage 4: bias + silu + atomic segment-sum scatter ----
  int dst0[8], dst1[8];
  #pragma unroll
  for (int r = 0; r < 8; ++r) {
    int g0 = e0 + r + (h << 3);
    int g1 = e0 + 16 + r + (h << 3);
    dst0[r] = (g0 < E) ? ei[g0] : -1;
    dst1[r] = (g1 < E) ? ei[g1] : -1;
  }
  #pragma unroll
  for (int nt = 0; nt < 8; ++nt) {
    float bias = b2[(nt << 4) + m];
    int   col  = (nt << 4) + m;
    #pragma unroll
    for (int r = 0; r < 8; ++r) {
      float v0 = silu_f(acc0[nt][r] + bias);
      float v1 = silu_f(acc1[nt][r] + bias);
      if (dst0[r] >= 0) atomicAdd(&agg[dst0[r] * 128 + col], v0);
      if (dst1[r] >= 0) atomicAdd(&agg[dst1[r] * 128 + col], v1);
    }
  }
}

// ---------------------------------------------------------------------------
// Node kernel: per wave, 32 nodes (2 M-tiles).
//   [x | agg] [32x192] -> silu(.@nw1+nb1) -> .@nw2 + nb2 + x
// ---------------------------------------------------------------------------
__global__ void __launch_bounds__(256)
node_mlp(const float* __restrict__ x,
         const float* __restrict__ agg,
         const unsigned short* __restrict__ w1f,  // 6x8 frags (global)
         const float* __restrict__ b1,
         const unsigned short* __restrict__ w2f,  // 4x4 frags (global)
         const float* __restrict__ b2,
         float* __restrict__ out,                 // [N][64]
         int N) {
  __shared__ __align__(16) unsigned short L[N_LDS];

  const int lane = threadIdx.x & 31;
  const int wv   = threadIdx.x >> 5;
  const int n0   = blockIdx.x * 256 + wv * 32;
  unsigned short* buf = &L[NS_OFF + wv * (2 * TILE_N)];

  const int m = lane & 15;
  const int h = lane >> 4;

  // ---- stage 0: stage weights into LDS ----
  {
    const u32x4* s1 = (const u32x4*)w1f;
    const u32x4* s2 = (const u32x4*)w2f;
    u32x4* d1 = (u32x4*)&L[NW1_OFF];
    u32x4* d2 = (u32x4*)&L[NW2_OFF];
    for (int i = threadIdx.x; i < 6 * 8 * 512 / 8; i += 256) d1[i] = s1[i];
    for (int i = threadIdx.x; i < 4 * 4 * 512 / 8; i += 256) d2[i] = s2[i];
  }

  // ---- stage 1: build [x | agg] tiles [32 x 192] (bf16 pairs) ----
  for (int e = 0; e < 32; ++e) {
    int node = n0 + e; if (node >= N) node = N - 1;
    unsigned* dst = (unsigned*)&buf[(e >> 4) * TILE_N + (e & 15) * LDN];
    #pragma unroll
    for (int j = 0; j < 3; ++j) {
      int p = (j << 5) + lane;          // 96 column pairs
      int c = p << 1;
      f32x2 t = (c < 64) ? *(const f32x2*)&x[node * 64 + c]
                         : *(const f32x2*)&agg[node * 128 + (c - 64)];
      dst[p] = pack2bf(t.x, t.y);
    }
  }

  __syncthreads();

  const unsigned short* wlane = &L[lane << 4];

  // ---- stage 2: layer 1 (K=192 -> 6 ktiles, N=128 -> 8 ntiles) ----
  v8f acc0[8], acc1[8];
  #pragma unroll
  for (int nt = 0; nt < 8; ++nt) { acc0[nt] = vzero8(); acc1[nt] = vzero8(); }
  #pragma unroll
  for (int kt = 0; kt < 6; ++kt) {
    v16bf a0 = frag_a_lds(&buf[m * LDN + kt * 32 + (h << 3)]);
    v16bf a1 = frag_a_lds(&buf[TILE_N + m * LDN + kt * 32 + (h << 3)]);
    v16bf bc = frag_b(wlane + NW1_OFF + (kt << 12));
    #pragma unroll
    for (int nt = 0; nt < 8; ++nt) {
      v16bf bn;
      if (nt < 7) bn = frag_b(wlane + NW1_OFF + (kt << 12) + ((nt + 1) << 9));
      acc0[nt] = __builtin_amdgcn_wmma_f32_16x16x32_bf16(
          false, a0, false, bc, (short)0, acc0[nt], false, false);
      acc1[nt] = __builtin_amdgcn_wmma_f32_16x16x32_bf16(
          false, a1, false, bc, (short)0, acc1[nt], false, false);
      if (nt < 7) bc = bn;
    }
  }
  // bias + silu -> hidden tiles, overlaying the input tiles
  #pragma unroll
  for (int nt = 0; nt < 8; ++nt) {
    float bias = b1[(nt << 4) + m];
    #pragma unroll
    for (int r = 0; r < 8; ++r) {
      float v0 = silu_f(acc0[nt][r] + bias);
      float v1 = silu_f(acc1[nt][r] + bias);
      buf[(r + (h << 3)) * LDH + (nt << 4) + m]          = f2bf(v0);
      buf[TILE_H + (r + (h << 3)) * LDH + (nt << 4) + m] = f2bf(v1);
    }
  }

  // ---- stage 3: layer 2 (K=128 -> 4 ktiles, N=64 -> 4 ntiles) ----
  v8f c20[4], c21[4];
  #pragma unroll
  for (int nt = 0; nt < 4; ++nt) { c20[nt] = vzero8(); c21[nt] = vzero8(); }
  #pragma unroll
  for (int kt = 0; kt < 4; ++kt) {
    v16bf a0 = frag_a_lds(&buf[m * LDH + kt * 32 + (h << 3)]);
    v16bf a1 = frag_a_lds(&buf[TILE_H + m * LDH + kt * 32 + (h << 3)]);
    v16bf bc = frag_b(wlane + NW2_OFF + (kt << 11));
    #pragma unroll
    for (int nt = 0; nt < 4; ++nt) {
      v16bf bn;
      if (nt < 3) bn = frag_b(wlane + NW2_OFF + (kt << 11) + ((nt + 1) << 9));
      c20[nt] = __builtin_amdgcn_wmma_f32_16x16x32_bf16(
          false, a0, false, bc, (short)0, c20[nt], false, false);
      c21[nt] = __builtin_amdgcn_wmma_f32_16x16x32_bf16(
          false, a1, false, bc, (short)0, c21[nt], false, false);
      if (nt < 3) bc = bn;
    }
  }

  // ---- epilogue: residual + bias, fp32 store ----
  #pragma unroll
  for (int nt = 0; nt < 4; ++nt) {
    float bias = b2[(nt << 4) + m];
    int   col  = (nt << 4) + m;
    #pragma unroll
    for (int r = 0; r < 8; ++r) {
      int n0r = n0 + r + (h << 3);
      int n1r = n0 + 16 + r + (h << 3);
      if (n0r < N) {
        float v = x[n0r * 64 + col] + c20[nt][r] + bias;
        __builtin_nontemporal_store(v, &out[n0r * 64 + col]);
      }
      if (n1r < N) {
        float v = x[n1r * 64 + col] + c21[nt][r] + bias;
        __builtin_nontemporal_store(v, &out[n1r * 64 + col]);
      }
    }
  }
}

// ---------------------------------------------------------------------------
extern "C" void kernel_launch(void* const* d_in, const int* in_sizes, int n_in,
                              void* d_out, int out_size, void* d_ws, size_t ws_size,
                              hipStream_t stream) {
  const float* x    = (const float*)d_in[0];
  const int*   ei   = (const int*)  d_in[1];
  const float* ewt  = (const float*)d_in[2];
  const float* ea   = (const float*)d_in[3];
  const float* ew1  = (const float*)d_in[4];
  const float* eb1  = (const float*)d_in[5];
  const float* ew2  = (const float*)d_in[6];
  const float* eb2  = (const float*)d_in[7];
  const float* nw1  = (const float*)d_in[8];
  const float* nb1  = (const float*)d_in[9];
  const float* nw2  = (const float*)d_in[10];
  const float* nb2  = (const float*)d_in[11];
  float* out = (float*)d_out;

  const int N = in_sizes[0] / 64;
  const int E = in_sizes[2];

  // Workspace layout
  char* ws = (char*)d_ws;
  size_t off = 0;
  auto align256 = [](size_t v) { return (v + 255) & ~(size_t)255; };
  float* agg = (float*)(ws + off);
  off = align256(off + (size_t)N * 128 * sizeof(float));
  unsigned short* w1f = (unsigned short*)(ws + off);  // 5x8 frags
  off = align256(off + (size_t)5 * 8 * 512 * 2);
  unsigned short* w2f = (unsigned short*)(ws + off);  // 4x8 frags
  off = align256(off + (size_t)4 * 8 * 512 * 2);
  unsigned short* n1f = (unsigned short*)(ws + off);  // 6x8 frags
  off = align256(off + (size_t)6 * 8 * 512 * 2);
  unsigned short* n2f = (unsigned short*)(ws + off);  // 4x4 frags
  off = align256(off + (size_t)4 * 4 * 512 * 2);

  // Pre-swizzle weights into WMMA-B fragment layout (bf16)
  swz_weights<<<80, 256, 0, stream>>>(ew1, w1f, 145, 5, 8, 128);
  swz_weights<<<64, 256, 0, stream>>>(ew2, w2f, 128, 4, 8, 128);
  swz_weights<<<96, 256, 0, stream>>>(nw1, n1f, 192, 6, 8, 128);
  swz_weights<<<32, 256, 0, stream>>>(nw2, n2f, 128, 4, 4, 64);

  // Zero the segment-sum accumulator (must happen every call)
  {
    int n4 = N * 128 / 4;
    zero_f32v4<<<(n4 + 255) / 256, 256, 0, stream>>>((float4*)agg, n4);
  }

  // Fused edge MLP + scatter (256 edges per block)
  {
    int grid = (E + 255) / 256;
    edge_mlp_scatter<<<grid, 256, 0, stream>>>(x, ei, ewt, ea, w1f, eb1, w2f,
                                               eb2, agg, E);
  }

  // Node MLP + residual (256 nodes per block)
  {
    int grid = (N + 255) / 256;
    node_mlp<<<grid, 256, 0, stream>>>(x, agg, n1f, nb1, n2f, nb2, out, N);
  }
}
